// GraphGatedMambaLayer_3547642986683
// MI455X (gfx1250) — compile-verified, matrix operand order checked by reading, not run on hardware
//
#include <hip/hip_runtime.h>

typedef __attribute__((ext_vector_type(16))) _Float16 v16h;
typedef __attribute__((ext_vector_type(8)))  _Float16 v8h;
typedef __attribute__((ext_vector_type(8)))  float    v8f;

#define DEV static __device__ __forceinline__

namespace {
constexpr int BB   = 4;
constexpr int TT   = 96;
constexpr int NODE = 128;
constexpr int DM   = 128;
constexpr int DIN  = 256;   // EXPAND*D_MODEL
constexpr int DST  = 16;
constexpr int DCONV= 4;
constexpr int DTR  = 8;
constexpr int DBCW = DTR + 2*DST; // 40
constexpr int BN   = BB*NODE;     // 512 sequences
constexpr int ROWS = BN*TT;       // 49152 (== B*T*NODE)
}

DEV float gelu_f(float x){ return 0.5f*x*(1.f+erff(x*0.70710678118654752f)); }
DEV float silu_f(float x){ return x/(1.f+__expf(-x)); }
DEV float sigmoid_f(float x){ return 1.f/(1.f+__expf(-x)); }
DEV float softplus_f(float x){ return (x>20.f)? x : log1pf(__expf(x)); }

// One lane's 16 halves of a 16x32 f16 WMMA operand: two contiguous 8-half runs
// (K = k0+g*8+{0..7} and +16) -> exactly two b128 loads, no conversion, and the
// packed halves land 2-per-VGPR in operand order.
DEV void load_frag_h(const _Float16* __restrict__ p, v16h& f){
  v8h lo = *(const v8h*)(p);
  v8h hi = *(const v8h*)(p + 16);
  #pragma unroll
  for (int h=0; h<8; h++){ f[h] = lo[h]; f[h+8] = hi[h]; }
}

// C = A(MxK,f16) @ Bw(NxK,f16)^T [+bias] [gelu]
// Each wave: 16 x (16*NT) tile; all K-step fragment loads are batched into
// independent registers before the NT back-to-back v_wmma_f32_16x16x32_f16.
// OUT: 0 = f32 store, 1 = f32 accumulate, 2 = f16 store (row-major, ldc),
//      3 = f16 store transposed within 128-row blocks (per-(b,t) msg^T)
template<int NT, int OUT, bool HASBIAS, bool GELU_ACT, bool NEDGE>
__global__ void __launch_bounds__(32)
gemm16h(const _Float16* __restrict__ A, const _Float16* __restrict__ Bw,
        const float* __restrict__ bias, void* __restrict__ Cv,
        int M, int N, int K, int lda, int ldb, int ldc,
        long long sA, long long sB, long long sC)
{
  A  += (long long)blockIdx.z * sA;
  Bw += (long long)blockIdx.z * sB;
  float*     Cf = (float*)Cv     + (long long)blockIdx.z * sC;
  _Float16*  Ch = (_Float16*)Cv  + (long long)blockIdx.z * sC;
  const int nb0 = blockIdx.x*(16*NT), mb = blockIdx.y*16;
  const int lane = threadIdx.x & 31, g = lane>>4, mr = lane&15;
  v8f acc[NT];
  #pragma unroll
  for (int t=0; t<NT; t++) acc[t] = (v8f){0.f,0.f,0.f,0.f,0.f,0.f,0.f,0.f};

  for (int k0=0; k0<K; k0+=32){
    v16h af, bf[NT];
    load_frag_h(A + (size_t)(mb+mr)*lda + k0 + g*8, af);
    #pragma unroll
    for (int t=0; t<NT; t++){
      const int bcol = nb0 + t*16 + mr;
      if (!NEDGE || bcol < N){
        load_frag_h(Bw + (size_t)bcol*ldb + k0 + g*8, bf[t]);
      } else {
        #pragma unroll
        for (int h=0; h<16; h++) bf[t][h] = (_Float16)0.f;
      }
    }
    #pragma unroll
    for (int t=0; t<NT; t++)
      acc[t] = __builtin_amdgcn_wmma_f32_16x16x32_f16(false, af, false, bf[t],
                                                      (short)0, acc[t], false, false);
  }

  #pragma unroll
  for (int t=0; t<NT; t++){
    const int bcol = nb0 + t*16 + mr;
    if (!NEDGE || bcol < N){
      float bv = HASBIAS ? bias[bcol] : 0.f;
      if (OUT == 3){
        v8h pk;
        #pragma unroll
        for (int r=0; r<8; r++){
          float v = acc[t][r] + bv;
          if (GELU_ACT) v = gelu_f(v);
          pk[r] = (_Float16)v;
        }
        int m0 = mb + g*8;     // 8 consecutive rows -> contiguous transposed run
        *(v8h*)(Ch + ((size_t)(m0>>7)*ldc + bcol)*NODE + (m0&127)) = pk;
      } else {
        #pragma unroll
        for (int r=0; r<8; r++){
          int m = mb + g*8 + r;
          float v = acc[t][r] + bv;
          if (GELU_ACT) v = gelu_f(v);
          size_t off = (size_t)m*ldc + bcol;
          if (OUT == 0)      Cf[off]  = v;
          else if (OUT == 1) Cf[off] += v;
          else               Ch[off]  = (_Float16)v;
        }
      }
    }
  }
}

__global__ void cvt_h_k(const float* __restrict__ in, _Float16* __restrict__ out, int n){
  int i = blockIdx.x*blockDim.x + threadIdx.x;
  if (i < n) out[i] = (_Float16)in[i];
}

__global__ void init_k(float* gsum){ if (threadIdx.x==0 && blockIdx.x==0) gsum[0]=0.f; }

__global__ void xmean_k(const float* __restrict__ x, float* __restrict__ xm){
  int idx = blockIdx.x*blockDim.x + threadIdx.x;
  if (idx >= BB*NODE*DM) return;
  int d = idx % DM, n = (idx/DM)%NODE, b = idx/(DM*NODE);
  float s = 0.f;
  for (int t=0; t<TT; t++) s += x[(((size_t)(b*TT+t))*NODE + n)*DM + d];
  xm[idx] = s * (1.f/TT);
}

__global__ void __launch_bounds__(32)
gate_k(const float* __restrict__ xm, const float* __restrict__ w1, const float* __restrict__ b1,
       const float* __restrict__ w2, const float* __restrict__ b2,
       float* __restrict__ gate, float* __restrict__ gsum){
  __shared__ float sh[32];
  int row = blockIdx.x, j = threadIdx.x;       // 32 hidden units
  const float* xr = xm + (size_t)row*DM;
  float acc = b1[j];
  for (int k=0; k<DM; k++) acc += xr[k]*w1[j*DM+k];
  sh[j] = gelu_f(acc)*w2[j];
  __syncthreads();
  for (int st=16; st>0; st>>=1){ if (j<st) sh[j]+=sh[j+st]; __syncthreads(); }
  if (j==0){ float gv = sigmoid_f(sh[0]+b2[0]); gate[row]=gv; atomicAdd(gsum, gv); }
}

__global__ void flag_k(const float* __restrict__ gsum, float* __restrict__ flag){
  flag[0] = (gsum[0]*(1.f/(BB*NODE)) > 0.01f) ? 1.f : 0.f;
}

__global__ void __launch_bounds__(128)
adj_k(const float* __restrict__ ne, _Float16* __restrict__ adjh){
  __shared__ float sh[128];
  int m = blockIdx.x, n = threadIdx.x;
  float e = 0.f;
  for (int k=0; k<16; k++) e += ne[m*16+k]*ne[n*16+k];
  sh[n]=e; __syncthreads();
  for (int st=64; st>0; st>>=1){ if(n<st) sh[n]=fmaxf(sh[n],sh[n+st]); __syncthreads(); }
  float mx = sh[0]; __syncthreads();
  float ex = __expf(e-mx);
  sh[n]=ex; __syncthreads();
  for (int st=64; st>0; st>>=1){ if(n<st) sh[n]+=sh[n+st]; __syncthreads(); }
  adjh[m*NODE+n] = (_Float16)(ex/sh[0]);
}

// LN(agg+x) -> gated blend, scattered into xf[(b*N+n), t, d]
__global__ void __launch_bounds__(128)
graphmix_k(const float* __restrict__ x, const float* __restrict__ agg,
           const float* __restrict__ gcg, const float* __restrict__ gcb,
           const float* __restrict__ gate, const float* __restrict__ flag,
           float* __restrict__ xf){
  __shared__ float sh[128];
  int d = threadIdx.x;
  int n = blockIdx.x % NODE;
  int t = (blockIdx.x / NODE) % TT;
  int b = blockIdx.x / (NODE*TT);
  size_t off = (((size_t)(b*TT+t))*NODE + n)*DM + d;
  float xv = x[off];
  float av = agg[off] + xv;
  sh[d]=av; __syncthreads();
  for (int st=64; st>0; st>>=1){ if(d<st) sh[d]+=sh[d+st]; __syncthreads(); }
  float mean = sh[0]*(1.f/DM); __syncthreads();
  float c = av-mean;
  sh[d]=c*c; __syncthreads();
  for (int st=64; st>0; st>>=1){ if(d<st) sh[d]+=sh[d+st]; __syncthreads(); }
  float var = sh[0]*(1.f/DM);
  float xg = c*rsqrtf(var+1e-5f)*gcg[d]+gcb[d];
  float gv = gate[b*NODE+n];
  float outv = (flag[0] > 0.5f) ? (xv + gv*(xg-xv)) : xv;
  xf[(((size_t)(b*NODE+n))*TT + t)*DM + d] = outv;
}

// LayerNorm(row of 128) -> f16 (consumed only by WMMA GEMMs)
__global__ void __launch_bounds__(128)
ln_k(const float* __restrict__ in, _Float16* __restrict__ out,
     const float* __restrict__ g, const float* __restrict__ bta){
  __shared__ float sh[128];
  int d = threadIdx.x; size_t row = blockIdx.x;
  float v = in[row*DM + d];
  sh[d]=v; __syncthreads();
  for (int st=64; st>0; st>>=1){ if(d<st) sh[d]+=sh[d+st]; __syncthreads(); }
  float mean = sh[0]*(1.f/DM); __syncthreads();
  float c = v-mean;
  sh[d]=c*c; __syncthreads();
  for (int st=64; st>0; st>>=1){ if(d<st) sh[d]+=sh[d+st]; __syncthreads(); }
  float var = sh[0]*(1.f/DM);
  out[row*DM+d] = (_Float16)(c*rsqrtf(var+1e-5f)*g[d]+bta[d]);
}

__global__ void conv_k(const float* __restrict__ xz, const float* __restrict__ cw,
                       const float* __restrict__ cb,
                       float* __restrict__ u, _Float16* __restrict__ uh){
  int idx = blockIdx.x*blockDim.x + threadIdx.x;
  if (idx >= ROWS*DIN) return;
  int c = idx % DIN; int r = idx / DIN; int t = r % TT; int bn = r / TT;
  float acc = cb[c];
  #pragma unroll
  for (int k=0; k<DCONV; k++){
    int tt = t - (DCONV-1) + k;
    if (tt >= 0) acc += xz[((size_t)(bn*TT+tt))*(2*DIN) + c] * cw[c*DCONV + k];
  }
  float v = silu_f(acc);
  u[idx] = v;
  uh[idx] = (_Float16)v;
}

__global__ void dt_k(const float* __restrict__ dbc, const float* __restrict__ dtw,
                     const float* __restrict__ dtb, float* __restrict__ dt){
  int idx = blockIdx.x*blockDim.x + threadIdx.x;
  if (idx >= ROWS*DIN) return;
  int c = idx % DIN; int r = idx / DIN;
  float acc = dtb[c];
  #pragma unroll
  for (int j=0; j<DTR; j++) acc += dbc[(size_t)r*DBCW + j]*dtw[c*DTR + j];
  dt[idx] = softplus_f(acc);
}

// Selective scan: one thread per (sequence bn, channel d); B/C broadcast via LDS.
// Emits y*silu(z) directly as f16 (only the out-proj WMMA consumes it).
__global__ void __launch_bounds__(256)
scan_k(const float* __restrict__ dt, const float* __restrict__ u,
       const float* __restrict__ dbc, const float* __restrict__ A_log,
       const float* __restrict__ Dp, const float* __restrict__ xz,
       _Float16* __restrict__ yh){
  __shared__ float sB[DST], sC[DST];
  int d = threadIdx.x, bn = blockIdx.x;
  float Av[DST], h[DST];
  #pragma unroll
  for (int s=0; s<DST; s++){ Av[s] = -__expf(A_log[d*DST+s]); h[s]=0.f; }
  float Dd = Dp[d];
  for (int t=0; t<TT; t++){
    int r = bn*TT + t;
    __syncthreads();
    if (d < 2*DST){
      float v = dbc[(size_t)r*DBCW + DTR + d];
      if (d < DST) sB[d]=v; else sC[d-DST]=v;
    }
    __syncthreads();
    float dtv = dt[(size_t)r*DIN + d];
    float uv  = u[(size_t)r*DIN + d];
    float du  = dtv*uv;
    float y = 0.f;
    #pragma unroll
    for (int s=0; s<DST; s++){
      h[s] = __expf(dtv*Av[s])*h[s] + du*sB[s];
      y += h[s]*sC[s];
    }
    float z = xz[(size_t)r*(2*DIN) + DIN + d];
    y = (y + uv*Dd) * silu_f(z);
    yh[(size_t)r*DIN + d] = (_Float16)y;
  }
}

__global__ void out_k(const float* __restrict__ xf, float* __restrict__ out){
  int idx = blockIdx.x*blockDim.x + threadIdx.x;
  if (idx >= BB*TT*NODE*DM) return;
  int d = idx % DM; int n = (idx/DM)%NODE; int t = (idx/(DM*NODE))%TT; int b = idx/(DM*NODE*TT);
  out[idx] = xf[(((size_t)(b*NODE+n))*TT + t)*DM + d];
}

extern "C" void kernel_launch(void* const* d_in, const int* in_sizes, int n_in,
                              void* d_out, int out_size, void* d_ws, size_t ws_size,
                              hipStream_t stream) {
  const float* x        = (const float*)d_in[0];
  const float* gate_w1  = (const float*)d_in[1];
  const float* gate_b1  = (const float*)d_in[2];
  const float* gate_w2  = (const float*)d_in[3];
  const float* gate_b2  = (const float*)d_in[4];
  const float* node_emb = (const float*)d_in[5];
  const float* msg_w    = (const float*)d_in[6];
  const float* msg_b    = (const float*)d_in[7];
  const float* gc_ln_g  = (const float*)d_in[8];
  const float* gc_ln_b  = (const float*)d_in[9];
  const float* n1_g     = (const float*)d_in[10];
  const float* n1_b     = (const float*)d_in[11];
  const float* in_w     = (const float*)d_in[12];
  const float* conv_w   = (const float*)d_in[13];
  const float* conv_b   = (const float*)d_in[14];
  const float* xproj_w  = (const float*)d_in[15];
  const float* dt_w     = (const float*)d_in[16];
  const float* dt_b     = (const float*)d_in[17];
  const float* A_log    = (const float*)d_in[18];
  const float* Dparam   = (const float*)d_in[19];
  const float* out_w    = (const float*)d_in[20];
  const float* n2_g     = (const float*)d_in[21];
  const float* n2_b     = (const float*)d_in[22];
  const float* ffn_w1   = (const float*)d_in[23];
  const float* ffn_b1   = (const float*)d_in[24];
  const float* ffn_w2   = (const float*)d_in[25];
  const float* ffn_b2   = (const float*)d_in[26];
  float* out = (float*)d_out;
  float* ws  = (float*)d_ws;

  // workspace layout (float units, all 16B aligned)
  const size_t o_xmean = 0;                                  // 65536
  const size_t o_gate  = o_xmean + (size_t)BB*NODE*DM;       // 512
  const size_t o_gsum  = o_gate + BN;                        // gsum+flag+pad (8)
  const size_t o_flag  = o_gsum + 1;
  const size_t o_adjh  = o_gsum + 8;                         // 16384 halves
  const size_t o_wh    = o_adjh + (size_t)NODE*NODE/2;       // 190464 halves pool
  const size_t o_xf    = o_wh   + 95232;                     // f32 residual (BN,T,D)
  const size_t o_xh    = o_xf   + (size_t)ROWS*DM;           // x in f16
  const size_t o_msgT  = o_xh   + (size_t)ROWS*DM/2;         // msg^T f16 per (b,t)
  const size_t o_agg   = o_msgT + (size_t)ROWS*DM/2;         // f32
  const size_t o_lnh   = o_agg  + (size_t)ROWS*DM;           // ln1/ln2 f16
  const size_t o_xz    = o_lnh  + (size_t)ROWS*DM/2;         // f32 (ROWS,512)
  const size_t o_u     = o_xz   + (size_t)ROWS*2*DIN;        // f32
  const size_t o_uh    = o_u    + (size_t)ROWS*DIN;          // u f16 / ffn hidden f16
  const size_t o_dbc   = o_uh   + (size_t)ROWS*DIN/2;        // f32
  const size_t o_dt    = o_dbc  + (size_t)ROWS*DBCW;         // f32
  const size_t o_yh    = o_dt   + (size_t)ROWS*DIN;          // y f16
  const size_t total_f = o_yh   + (size_t)ROWS*DIN/2;
  if (ws_size < total_f*sizeof(float)) return;

  float*     xmean = ws + o_xmean;
  float*     gate  = ws + o_gate;
  float*     gsum  = ws + o_gsum;
  float*     flag  = ws + o_flag;
  _Float16*  adjh  = (_Float16*)(ws + o_adjh);
  _Float16*  whp   = (_Float16*)(ws + o_wh);
  _Float16*  wh_msg   = whp;            // 16384
  _Float16*  wh_in    = whp + 16384;    // 65536
  _Float16*  wh_xproj = whp + 81920;    // 10240
  _Float16*  wh_out   = whp + 92160;    // 32768
  _Float16*  wh_ffn1  = whp + 124928;   // 32768
  _Float16*  wh_ffn2  = whp + 157696;   // 32768
  float*     xf    = ws + o_xf;
  _Float16*  xh    = (_Float16*)(ws + o_xh);
  _Float16*  msgT  = (_Float16*)(ws + o_msgT);
  float*     agg   = ws + o_agg;
  _Float16*  lnh   = (_Float16*)(ws + o_lnh);
  float*     xz    = ws + o_xz;
  float*     ubuf  = ws + o_u;
  _Float16*  uh    = (_Float16*)(ws + o_uh);   // later: FFN hidden f16
  float*     dbc   = ws + o_dbc;
  float*     dtb_  = ws + o_dt;
  _Float16*  yh    = (_Float16*)(ws + o_yh);

  // ---- one-time f16 conversions (weights + input) ----
  cvt_h_k<<<(DM*DM+255)/256, 256, 0, stream>>>(msg_w, wh_msg, DM*DM);
  cvt_h_k<<<(2*DIN*DM+255)/256, 256, 0, stream>>>(in_w, wh_in, 2*DIN*DM);
  cvt_h_k<<<(DBCW*DIN+255)/256, 256, 0, stream>>>(xproj_w, wh_xproj, DBCW*DIN);
  cvt_h_k<<<(DM*DIN+255)/256, 256, 0, stream>>>(out_w, wh_out, DM*DIN);
  cvt_h_k<<<(2*DM*DM+255)/256, 256, 0, stream>>>(ffn_w1, wh_ffn1, 2*DM*DM);
  cvt_h_k<<<(DM*2*DM+255)/256, 256, 0, stream>>>(ffn_w2, wh_ffn2, DM*2*DM);
  cvt_h_k<<<(ROWS*DM+255)/256, 256, 0, stream>>>(x, xh, ROWS*DM);

  // ---- gate path ----
  init_k<<<1, 1, 0, stream>>>(gsum);
  xmean_k<<<(BB*NODE*DM+255)/256, 256, 0, stream>>>(x, xmean);
  gate_k<<<BN, 32, 0, stream>>>(xmean, gate_w1, gate_b1, gate_w2, gate_b2, gate, gsum);
  flag_k<<<1, 1, 0, stream>>>(gsum, flag);
  adj_k<<<NODE, 128, 0, stream>>>(node_emb, adjh);

  // ---- graph mixing ----
  // msg^T(f16, per-(b,t) 128x128 blocks) = (x @ msg_w^T + msg_b)^T
  gemm16h<4,3,true,false,false><<<dim3(DM/64, ROWS/16, 1), 32, 0, stream>>>(
      xh, wh_msg, msg_b, msgT, ROWS, DM, DM, DM, DM, DM, 0, 0, 0);
  // agg[bt] = adj @ msg[bt]  ==  adj(MxK) @ msgT(NxK)^T   (batched WMMA)
  gemm16h<4,0,false,false,false><<<dim3(DM/64, NODE/16, BB*TT), 32, 0, stream>>>(
      adjh, msgT, nullptr, agg, NODE, DM, NODE, NODE, NODE, DM,
      0LL, (long long)DM*NODE, (long long)NODE*DM);
  graphmix_k<<<BB*TT*NODE, 128, 0, stream>>>(x, agg, gc_ln_g, gc_ln_b, gate, flag, xf);

  // ---- Mamba block ----
  ln_k<<<ROWS, 128, 0, stream>>>(xf, lnh, n1_g, n1_b);
  gemm16h<4,0,false,false,false><<<dim3((2*DIN)/64, ROWS/16, 1), 32, 0, stream>>>(
      lnh, wh_in, nullptr, xz, ROWS, 2*DIN, DM, DM, DM, 2*DIN, 0, 0, 0);
  conv_k<<<(ROWS*DIN+255)/256, 256, 0, stream>>>(xz, conv_w, conv_b, ubuf, uh);
  gemm16h<1,0,false,false,true><<<dim3((DBCW+15)/16, ROWS/16, 1), 32, 0, stream>>>(
      uh, wh_xproj, nullptr, dbc, ROWS, DBCW, DIN, DIN, DIN, DBCW, 0, 0, 0);
  dt_k<<<(ROWS*DIN+255)/256, 256, 0, stream>>>(dbc, dt_w, dt_b, dtb_);
  scan_k<<<BN, DIN, 0, stream>>>(dtb_, ubuf, dbc, A_log, Dparam, xz, yh);
  // xf += y @ out_w^T
  gemm16h<4,1,false,false,false><<<dim3(DM/64, ROWS/16, 1), 32, 0, stream>>>(
      yh, wh_out, nullptr, xf, ROWS, DM, DIN, DIN, DIN, DM, 0, 0, 0);

  // ---- FFN ----
  ln_k<<<ROWS, 128, 0, stream>>>(xf, lnh, n2_g, n2_b);
  gemm16h<4,2,true,true,false><<<dim3((2*DM)/64, ROWS/16, 1), 32, 0, stream>>>(
      lnh, wh_ffn1, ffn_b1, uh, ROWS, 2*DM, DM, DM, DM, 2*DM, 0, 0, 0);
  gemm16h<4,1,true,false,false><<<dim3(DM/64, ROWS/16, 1), 32, 0, stream>>>(
      uh, wh_ffn2, ffn_b2, xf, ROWS, DM, 2*DM, 2*DM, 2*DM, DM, 0, 0, 0);

  // ---- transpose back to (B,T,N,D) ----
  out_k<<<(BB*TT*NODE*DM+255)/256, 256, 0, stream>>>(xf, out);
}
